// DeepseekV4Indexer_50826642981096
// MI455X (gfx1250) — matrix operand order repaired and checked by reading, not compiled, not run on hardware
//
#include <hip/hip_runtime.h>
#include <hip/hip_bf16.h>

// ---------------------------------------------------------------------------
// DeepseekV4 indexer for MI455X (gfx1250, wave32, WMMA).
// Heavy math: v_wmma_f32_16x16x32_bf16 (f32 accumulation).
// Tile staging: Tensor Data Mover (TENSOR_LOAD_TO_LDS, TENSORcnt) with
// double-buffered LDS, plus GLOBAL_LOAD_ASYNC_TO_LDS (ASYNCcnt) for A tiles
// in the q kernel.  CDNA5 ISA-8.
// ---------------------------------------------------------------------------

typedef __attribute__((ext_vector_type(16))) __bf16        v16bf;
typedef __attribute__((ext_vector_type(8)))  float         v8f;
typedef __attribute__((ext_vector_type(4)))  unsigned int  u32x4;
typedef __attribute__((ext_vector_type(8)))  int           i32x8;
typedef __attribute__((ext_vector_type(4)))  int           i32x4;

static constexpr int kS  = 4096;   // sequence
static constexpr int kH  = 4096;   // hidden
static constexpr int kD  = 128;    // head dim
static constexpr int kNH = 64;     // num heads
static constexpr int kQ  = 1024;   // q residual dim
static constexpr int kP  = 1024;   // pooled length (S / RATIO)
static constexpr int kTopK = 512;
static constexpr float kTheta = 160000.0f;

// ---------------- helpers ----------------

__device__ __forceinline__ unsigned short f2bf(float f) {
  unsigned int u = __float_as_uint(f);
  u += 0x7fffu + ((u >> 16) & 1u);          // round-to-nearest-even
  return (unsigned short)(u >> 16);
}

// LDS byte offset from a generic pointer to a __shared__ object.
// addrspacecast(3->flat) = {aperture_hi32, lds_offset32}.
__device__ __forceinline__ unsigned lds_off(const void* p) {
  return (unsigned)(size_t)p;
}

// --- Tensor Data Mover: 2D tile load, tensor dims == tile dims (no OOB),
// data_size = 2 bytes, row pitch in elements.  D# per ISA §8.3/8.4. ---
__device__ __forceinline__ void tdm_load_2d(unsigned lds_addr, const void* g,
                                            unsigned cols, unsigned rows,
                                            unsigned pitch_elems) {
  unsigned long long ga = (unsigned long long)(size_t)g;
  u32x4 g0 = { 1u,                                   // count=1 (valid user D#)
               lds_addr,                             // lds_addr [63:32]
               (unsigned)ga,                         // global_addr[95:64]
               (unsigned)((ga >> 32) & 0x01ffffffu) | 0x80000000u }; // +type=2
  i32x8 g1 = { (int)(1u << 16),                      // data_size=1 -> 2 bytes
               (int)((cols & 0xffffu) << 16),        // tensor_dim0[15:0]
               (int)((cols >> 16) | ((rows & 0xffffu) << 16)),  // td0hi|td1lo
               (int)((rows >> 16) | ((cols & 0xffffu) << 16)),  // td1hi|tile0
               (int)(rows & 0xffffu),                // tile_dim1 (tile_dim2=0)
               (int)pitch_elems,                     // tensor_dim0_stride lo
               0, 0 };
  i32x4 z4 = { 0, 0, 0, 0 };
#if __clang_major__ >= 23
  i32x8 z8 = { 0, 0, 0, 0, 0, 0, 0, 0 };
  __builtin_amdgcn_tensor_load_to_lds(g0, g1, z4, z4, z8, 0);
#else
  __builtin_amdgcn_tensor_load_to_lds(g0, g1, z4, z4, 0);
#endif
}

__device__ __forceinline__ void wait_tensor0() {
  __builtin_amdgcn_s_wait_tensorcnt(0);
}

// GLOBAL_LOAD_ASYNC_TO_LDS (GV form), tracked with ASYNCcnt.
__device__ __forceinline__ void async_load_b64(unsigned lds, const void* g) {
  asm volatile("global_load_async_to_lds_b64 %0, %1, off"
               :: "v"(lds), "v"(g) : "memory");
}
__device__ __forceinline__ void wait_async0() {
  asm volatile("s_wait_asynccnt 0" ::: "memory");
}

union Frag { v16bf v; u32x4 q[2]; };

// A tile in LDS as [16][32] bf16, K contiguous (lane half h holds K=16h..16h+15, m = lane%16).
__device__ __forceinline__ v16bf load_a_frag(const unsigned short* A) {
  int l = threadIdx.x & 31;
  int m = l & 15, h = l >> 4;
  Frag f;
  f.q[0] = *(const u32x4*)(A + m * 32 + 16 * h);
  f.q[1] = *(const u32x4*)(A + m * 32 + 16 * h + 8);
  return f.v;
}

// B tile in LDS as [n][k] = [16][32] bf16, K contiguous
// (lanes 0-15 hold K=0..15, lanes 16-31 hold K=16..31, n = lane%16).
__device__ __forceinline__ v16bf load_b_frag(const unsigned short* B) {
  int l = threadIdx.x & 31;
  int n = l & 15, h = l >> 4;
  Frag f;
  f.q[0] = *(const u32x4*)(B + n * 32 + 16 * h);
  f.q[1] = *(const u32x4*)(B + n * 32 + 16 * h + 8);
  return f.v;
}

__device__ __forceinline__ v8f wmma_bf16(v16bf a, v16bf b, v8f c) {
  return __builtin_amdgcn_wmma_f32_16x16x32_bf16(
      false, a, false, b, (short)0, c, false, false);
}

// ---------------- elementwise converts ----------------

__global__ void f32_to_bf16_kernel(const float* __restrict__ src,
                                   unsigned short* __restrict__ dst, int n) {
  int i = blockIdx.x * blockDim.x + threadIdx.x;
  if (i < n) dst[i] = f2bf(src[i]);
}

// cos/sin table for positions 0..S-1, 64 rope dims (concat([f,f]) layout).
__global__ void rope_table_kernel(float* __restrict__ cos_t,
                                  float* __restrict__ sin_t) {
  int i = blockIdx.x * blockDim.x + threadIdx.x;
  if (i >= kS * 64) return;
  int pos = i >> 6;
  int d = i & 63;
  int j = d & 31;                              // arange(0,64,2)/64 = j/32
  float inv = powf(kTheta, -(float)j / 32.0f);
  float ang = (float)pos * inv;
  cos_t[i] = cosf(ang);
  sin_t[i] = sinf(ang);
}

// ---------------- fused kv/gate GEMM + pool + rmsnorm + rope ----------------
// Block: 32 S rows (= 8 pooled rows), 256 threads = 8 waves.
// Double-buffered tiles via TDM; epilogue f32 scratch aliases dead tile bufs.
__global__ __launch_bounds__(256) void pool_kernel(
    const unsigned short* __restrict__ hs_bf,     // [S][H]
    const unsigned short* __restrict__ wkv_bf,    // [D][H]
    const unsigned short* __restrict__ wgate_bf,  // [D][H]
    const float* __restrict__ ape,                // [4][128]
    const float* __restrict__ kv_norm_w,          // [128]
    const float* __restrict__ cos_t,
    const float* __restrict__ sin_t,
    unsigned short* __restrict__ pooled_bf)       // [P][128]
{
  // tiles: A[2] @0/2048 (32x32 bf16), B[2] @4096/20480 (256x32 bf16)
  // epilogue alias: kv_s @0, gate_s @16384, pool_s @32768
  __shared__ __align__(16) char smem[36864];

  const int tid = threadIdx.x;
  const int wave = tid >> 5, lane = tid & 31;
  const int s0 = blockIdx.x * 32;
  const int mt = wave >> 2;           // 0..1
  const int nt0 = (wave & 3) * 4;     // 4 of 16 ntiles
  const unsigned base = lds_off(smem);

  auto issue = [&](int kk, int b) {
    tdm_load_2d(base + b * 2048, hs_bf + (size_t)s0 * kH + kk, 32, 32, kH);
    unsigned bb = base + 4096 + b * 16384;
    tdm_load_2d(bb,        wkv_bf   + kk, 32, 128, kH);
    tdm_load_2d(bb + 8192, wgate_bf + kk, 32, 128, kH);
  };

  v8f acc[4] = {};
  if (wave == 0) { issue(0, 0); wait_tensor0(); }
  __syncthreads();
  int cur = 0;
  for (int k0 = 0; k0 < kH; k0 += 32) {
    if (wave == 0 && k0 + 32 < kH) issue(k0 + 32, cur ^ 1);
    const unsigned short* A = (const unsigned short*)(smem + cur * 2048);
    const unsigned short* B = (const unsigned short*)(smem + 4096 + cur * 16384);
    v16bf a = load_a_frag(A + mt * 16 * 32);
#pragma unroll
    for (int t = 0; t < 4; ++t) {
      v16bf b = load_b_frag(B + (nt0 + t) * 16 * 32);
      acc[t] = wmma_bf16(a, b, acc[t]);
    }
    if (wave == 0) wait_tensor0();
    __syncthreads();
    cur ^= 1;
  }
  float (*kv_s)[128]   = (float (*)[128])(smem);
  float (*gate_s)[128] = (float (*)[128])(smem + 16384);
  float (*pool_s)[128] = (float (*)[128])(smem + 32768);
  // dump accumulators (C layout: M = v + 8*(lane>>4), N = lane&15)
#pragma unroll
  for (int t = 0; t < 4; ++t) {
    int ntile = nt0 + t;
    int col = (ntile & 7) * 16 + (lane & 15);
    bool isGate = ntile >= 8;
#pragma unroll
    for (int v = 0; v < 8; ++v) {
      int row = mt * 16 + v + 8 * (lane >> 4);
      if (isGate) gate_s[row][col] = acc[t][v];
      else        kv_s[row][col]   = acc[t][v];
    }
  }
  __syncthreads();
  // softmax(gate+ape) pooling over RATIO=4
  for (int item = tid; item < 1024; item += 256) {
    int pl = item >> 7, d = item & 127;
    float g0 = gate_s[pl * 4 + 0][d] + ape[0 * 128 + d];
    float g1 = gate_s[pl * 4 + 1][d] + ape[1 * 128 + d];
    float g2 = gate_s[pl * 4 + 2][d] + ape[2 * 128 + d];
    float g3 = gate_s[pl * 4 + 3][d] + ape[3 * 128 + d];
    float m = fmaxf(fmaxf(g0, g1), fmaxf(g2, g3));
    float e0 = __expf(g0 - m), e1 = __expf(g1 - m);
    float e2 = __expf(g2 - m), e3 = __expf(g3 - m);
    float inv = 1.0f / (e0 + e1 + e2 + e3);
    pool_s[pl][d] = (kv_s[pl * 4 + 0][d] * e0 + kv_s[pl * 4 + 1][d] * e1 +
                     kv_s[pl * 4 + 2][d] * e2 + kv_s[pl * 4 + 3][d] * e3) * inv;
  }
  __syncthreads();
  // rmsnorm + partial rope; wave w owns pooled row w
  {
    int row = wave;
    float ss = 0.f;
#pragma unroll
    for (int i = 0; i < 4; ++i) {
      float x = pool_s[row][lane + 32 * i];
      ss += x * x;
    }
#pragma unroll
    for (int off = 16; off > 0; off >>= 1) ss += __shfl_xor(ss, off, 32);
    float rms = rsqrtf(ss / 128.0f + 1e-6f);
    int p = blockIdx.x * 8 + row;
    int pos = p * 4;  // RATIO
#pragma unroll
    for (int i = 0; i < 4; ++i) {
      int d = lane + 32 * i;
      float x = pool_s[row][d] * rms * kv_norm_w[d];
      float out;
      if (d < 64) {
        out = x;
      } else {
        int j = d - 64;
        int od = (j < 32) ? d + 32 : d - 32;
        float other = pool_s[row][od] * rms * kv_norm_w[od];
        float c = cos_t[pos * 64 + j], sn = sin_t[pos * 64 + j];
        out = x * c + ((j < 32) ? -other : other) * sn;
      }
      pooled_bf[(size_t)p * 128 + d] = f2bf(out);
    }
  }
}

// ---------------- wts = hidden @ weights_proj.T * NH^-0.5 ----------------
__global__ __launch_bounds__(256) void wts_kernel(
    const unsigned short* __restrict__ hs_bf,     // [S][H]
    const unsigned short* __restrict__ wproj_bf,  // [NH][H]
    float* __restrict__ wts)                      // [S][NH]
{
  __shared__ __align__(16) unsigned short tA[2][32 * 32];
  __shared__ __align__(16) unsigned short tB[2][64 * 32];
  const int tid = threadIdx.x, wave = tid >> 5, lane = tid & 31;
  const int s0 = blockIdx.x * 32;
  const int mt = wave >> 2, nt = wave & 3;

  auto issue = [&](int kk, int b) {
    tdm_load_2d(lds_off(tA[b]), hs_bf + (size_t)s0 * kH + kk, 32, 32, kH);
    tdm_load_2d(lds_off(tB[b]), wproj_bf + kk, 32, 64, kH);
  };

  v8f acc = {};
  if (wave == 0) { issue(0, 0); wait_tensor0(); }
  __syncthreads();
  int cur = 0;
  for (int k0 = 0; k0 < kH; k0 += 32) {
    if (wave == 0 && k0 + 32 < kH) issue(k0 + 32, cur ^ 1);
    v16bf a = load_a_frag(tA[cur] + mt * 16 * 32);
    v16bf b = load_b_frag(tB[cur] + nt * 16 * 32);
    acc = wmma_bf16(a, b, acc);
    if (wave == 0) wait_tensor0();
    __syncthreads();
    cur ^= 1;
  }
  const float scale = 0.125f;  // 64^-0.5
#pragma unroll
  for (int v = 0; v < 8; ++v) {
    int row = mt * 16 + v + 8 * (lane >> 4);
    int col = nt * 16 + (lane & 15);
    wts[(size_t)(s0 + row) * kNH + col] = acc[v] * scale;
  }
}

// ---------------- q = q_res @ wq_b.T, partial rope ----------------
// grid (S/32, NH).  A tiles via ASYNCcnt loads, B tiles via TDM.
__global__ __launch_bounds__(256) void q_kernel(
    const unsigned short* __restrict__ qres_bf,  // [S][Q]
    const unsigned short* __restrict__ wqb_bf,   // [NH*D][Q]
    const float* __restrict__ cos_t,
    const float* __restrict__ sin_t,
    unsigned short* __restrict__ q_bf)           // [S][NH][D]
{
  // tiles: A[2] @0/2048 (32x32), B[2] @4096/12288 (128x32); q_s alias @0
  __shared__ __align__(16) char smem[20480];
  const int tid = threadIdx.x, wave = tid >> 5, lane = tid & 31;
  const int s0 = blockIdx.x * 32;
  const int head = blockIdx.y;
  const int mt = wave >> 2, nt0 = (wave & 3) * 2;
  const unsigned base = lds_off(smem);

  auto issueA = [&](int kk, int b) {  // all 256 threads: async b64 each
    int row = tid >> 3, part = (tid & 7) * 4;
    async_load_b64(base + b * 2048 + row * 64 + part * 2,
                   qres_bf + (size_t)(s0 + row) * kQ + kk + part);
  };
  auto issueB = [&](int kk, int b) {  // wave 0: one TDM descriptor
    tdm_load_2d(base + 4096 + b * 8192,
                wqb_bf + (size_t)(head * 128) * kQ + kk, 32, 128, kQ);
  };

  v8f acc[2] = {};
  issueA(0, 0);
  if (wave == 0) issueB(0, 0);
  wait_async0();
  if (wave == 0) wait_tensor0();
  __syncthreads();
  int cur = 0;
  for (int k0 = 0; k0 < kQ; k0 += 32) {
    if (k0 + 32 < kQ) {
      issueA(k0 + 32, cur ^ 1);
      if (wave == 0) issueB(k0 + 32, cur ^ 1);
    }
    const unsigned short* A = (const unsigned short*)(smem + cur * 2048);
    const unsigned short* B = (const unsigned short*)(smem + 4096 + cur * 8192);
    v16bf a = load_a_frag(A + mt * 16 * 32);
#pragma unroll
    for (int t = 0; t < 2; ++t) {
      v16bf b = load_b_frag(B + (nt0 + t) * 16 * 32);
      acc[t] = wmma_bf16(a, b, acc[t]);
    }
    wait_async0();
    if (wave == 0) wait_tensor0();
    __syncthreads();
    cur ^= 1;
  }
  float (*q_s)[128] = (float (*)[128])(smem);
#pragma unroll
  for (int t = 0; t < 2; ++t) {
    int col = (nt0 + t) * 16 + (lane & 15);
#pragma unroll
    for (int v = 0; v < 8; ++v) {
      int row = mt * 16 + v + 8 * (lane >> 4);
      q_s[row][col] = acc[t][v];
    }
  }
  __syncthreads();
  // rope epilogue: wave owns 4 rows
  for (int r = 0; r < 4; ++r) {
    int row = wave * 4 + r;
    int s = s0 + row;
#pragma unroll
    for (int i = 0; i < 4; ++i) {
      int d = lane + 32 * i;
      float x = q_s[row][d];
      float out;
      if (d < 64) {
        out = x;
      } else {
        int j = d - 64;
        float other = q_s[row][(j < 32) ? d + 32 : d - 32];
        float c = cos_t[s * 64 + j], sn = sin_t[s * 64 + j];
        out = x * c + ((j < 32) ? -other : other) * sn;
      }
      q_bf[((size_t)s * kNH + head) * kD + d] = f2bf(out);
    }
  }
}

// ---------------- fused scores + head reduction ----------------
// grid (S, P/256). Block: heads 0..63 x 256 pooled cols; reduce over heads
// with wts in the epilogue so (S,NH,P) is never materialized.
__global__ __launch_bounds__(256) void score_kernel(
    const unsigned short* __restrict__ q_bf,      // [S][NH][D]
    const unsigned short* __restrict__ pooled_bf, // [P][D]
    const float* __restrict__ wts,                // [S][NH]
    float* __restrict__ index_scores)             // [S][P]
{
  // tiles: A[2] @0/4096 (64x32), B[2] @8192/24576 (256x32)
  __shared__ __align__(16) char smem[40960];
  __shared__ float wts_s[64];
  const int tid = threadIdx.x, wave = tid >> 5, lane = tid & 31;
  const int s = blockIdx.x;
  const int p0 = blockIdx.y * 256;
  const int nt0 = wave * 2;
  const unsigned base = lds_off(smem);
  if (tid < 64) wts_s[tid] = wts[(size_t)s * kNH + tid];

  auto issue = [&](int kk, int b) {
    tdm_load_2d(base + b * 4096, q_bf + (size_t)s * kNH * kD + kk, 32, 64, kD);
    tdm_load_2d(base + 8192 + b * 16384,
                pooled_bf + (size_t)p0 * kD + kk, 32, 256, kD);
  };

  v8f acc[2][4] = {};
  if (wave == 0) { issue(0, 0); wait_tensor0(); }
  __syncthreads();
  int cur = 0;
  for (int k0 = 0; k0 < kD; k0 += 32) {
    if (wave == 0 && k0 + 32 < kD) issue(k0 + 32, cur ^ 1);
    const unsigned short* A = (const unsigned short*)(smem + cur * 4096);
    const unsigned short* B = (const unsigned short*)(smem + 8192 + cur * 16384);
#pragma unroll
    for (int t = 0; t < 2; ++t) {
      v16bf b = load_b_frag(B + (nt0 + t) * 16 * 32);
#pragma unroll
      for (int m = 0; m < 4; ++m) {
        v16bf a = load_a_frag(A + m * 16 * 32);
        acc[t][m] = wmma_bf16(a, b, acc[t][m]);
      }
    }
    if (wave == 0) wait_tensor0();
    __syncthreads();
    cur ^= 1;
  }
  const float dscale = 0.08838834764831845f;  // 128^-0.5
#pragma unroll
  for (int t = 0; t < 2; ++t) {
    float part = 0.f;
#pragma unroll
    for (int m = 0; m < 4; ++m) {
#pragma unroll
      for (int v = 0; v < 8; ++v) {
        int h = m * 16 + v + 8 * (lane >> 4);
        part += fmaxf(acc[t][m][v], 0.f) * dscale * wts_s[h];
      }
    }
    part += __shfl_xor(part, 16, 32);  // combine the two M-halves (same N)
    if (lane < 16) {
      int col = p0 + (nt0 + t) * 16 + lane;
      index_scores[(size_t)s * kP + col] = part;
    }
  }
}

// ---------------- top-k (bitonic sort of 1024, take first 512 idx) --------
__global__ __launch_bounds__(256) void topk_kernel(
    const float* __restrict__ index_scores, int* __restrict__ out) {
  __shared__ float key[1024];
  __shared__ int   idx[1024];
  const int tid = threadIdx.x;
  const int s = blockIdx.x;
  for (int i = tid; i < 1024; i += 256) {
    key[i] = index_scores[(size_t)s * kP + i];
    idx[i] = i;
  }
  __syncthreads();
  for (int k = 2; k <= 1024; k <<= 1) {
    for (int j = k >> 1; j > 0; j >>= 1) {
      for (int i = tid; i < 1024; i += 256) {
        int ixj = i ^ j;
        if (ixj > i) {
          bool descend = ((i & k) == 0);
          float a = key[i], b = key[ixj];
          int ia = idx[i], ib = idx[ixj];
          bool aFirst = (a > b) || (a == b && ia < ib);
          if (aFirst != descend) {
            key[i] = b; key[ixj] = a;
            idx[i] = ib; idx[ixj] = ia;
          }
        }
      }
      __syncthreads();
    }
  }
  for (int i = tid; i < kTopK; i += 256) out[(size_t)s * kTopK + i] = idx[i];
}

// ---------------- launcher ----------------

extern "C" void kernel_launch(void* const* d_in, const int* in_sizes, int n_in,
                              void* d_out, int out_size, void* d_ws,
                              size_t ws_size, hipStream_t stream) {
  (void)in_sizes; (void)n_in; (void)out_size; (void)ws_size;
  const float* hidden = (const float*)d_in[0];  // [1,S,H]
  const float* q_res  = (const float*)d_in[1];  // [1,S,Q]
  const float* wkv    = (const float*)d_in[2];  // [D,H]
  const float* wgate  = (const float*)d_in[3];  // [D,H]
  const float* ape    = (const float*)d_in[4];  // [4,D]
  const float* kvnw   = (const float*)d_in[5];  // [D]
  const float* wq_b   = (const float*)d_in[6];  // [NH*D,Q]
  const float* wproj  = (const float*)d_in[7];  // [NH,H]

  char* ws = (char*)d_ws;
  size_t off = 0;
  auto carve = [&](size_t bytes) {
    char* p = ws + off;
    off += (bytes + 255) & ~(size_t)255;
    return p;
  };
  unsigned short* hs_bf    = (unsigned short*)carve((size_t)kS * kH * 2);
  unsigned short* qres_bf  = (unsigned short*)carve((size_t)kS * kQ * 2);
  unsigned short* wqb_bf   = (unsigned short*)carve((size_t)kNH * kD * kQ * 2);
  unsigned short* wkv_bf   = (unsigned short*)carve((size_t)kD * kH * 2);
  unsigned short* wgate_bf = (unsigned short*)carve((size_t)kD * kH * 2);
  unsigned short* wproj_bf = (unsigned short*)carve((size_t)kNH * kH * 2);
  float*          cos_t    = (float*)carve((size_t)kS * 64 * 4);
  float*          sin_t    = (float*)carve((size_t)kS * 64 * 4);
  unsigned short* pooledbf = (unsigned short*)carve((size_t)kP * kD * 2);
  unsigned short* q_bf     = (unsigned short*)carve((size_t)kS * kNH * kD * 2);
  float*          wts_buf  = (float*)carve((size_t)kS * kNH * 4);
  float*          isc      = (float*)carve((size_t)kS * kP * 4);

  auto cvt = [&](const float* s, unsigned short* d, long long n) {
    f32_to_bf16_kernel<<<(int)((n + 255) / 256), 256, 0, stream>>>(s, d, (int)n);
  };
  cvt(hidden, hs_bf,   (long long)kS * kH);
  cvt(q_res,  qres_bf, (long long)kS * kQ);
  cvt(wq_b,   wqb_bf,  (long long)kNH * kD * kQ);
  cvt(wkv,    wkv_bf,  (long long)kD * kH);
  cvt(wgate,  wgate_bf,(long long)kD * kH);
  cvt(wproj,  wproj_bf,(long long)kNH * kH);

  rope_table_kernel<<<(kS * 64 + 255) / 256, 256, 0, stream>>>(cos_t, sin_t);

  pool_kernel<<<kP / 8, 256, 0, stream>>>(hs_bf, wkv_bf, wgate_bf, ape, kvnw,
                                          cos_t, sin_t, pooledbf);
  wts_kernel<<<kS / 32, 256, 0, stream>>>(hs_bf, wproj_bf, wts_buf);
  q_kernel<<<dim3(kS / 32, kNH), 256, 0, stream>>>(qres_bf, wqb_bf, cos_t,
                                                   sin_t, q_bf);
  score_kernel<<<dim3(kS, kP / 256), 256, 0, stream>>>(q_bf, pooledbf, wts_buf,
                                                       isc);
  topk_kernel<<<kS, 256, 0, stream>>>(isc, (int*)d_out);
}
